// GAT_87076166960134
// MI455X (gfx1250) — matrix operand order repaired and checked by reading, not compiled, not run on hardware
//
#include <hip/hip_runtime.h>
#include <hip/hip_bf16.h>

// ---------------------------------------------------------------------------
// GAT 3-layer pipeline for MI455X (gfx1250, wave32).
// GEMMs via v_wmma_f32_16x16x32_bf16 (bf16 in, f32 acc); edge softmax +
// aggregation via native f32/u32 global atomics. 192MB L2 holds the whole
// activation matrix, so WMMA fragments stream directly from global memory.
// ---------------------------------------------------------------------------

typedef __attribute__((ext_vector_type(16))) __bf16 v16bf;
typedef __attribute__((ext_vector_type(8)))  __bf16 v8bf;
typedef __attribute__((ext_vector_type(8)))  float  v8f;

// order-preserving encode for float -> uint atomicMax
__device__ __forceinline__ unsigned enc_f32(float f) {
    unsigned u = __float_as_uint(f);
    return (u & 0x80000000u) ? ~u : (u | 0x80000000u);
}
__device__ __forceinline__ float dec_f32(unsigned u) {
    return (u & 0x80000000u) ? __uint_as_float(u & 0x7fffffffu)
                             : __uint_as_float(~u);
}
__device__ __forceinline__ unsigned short f32_to_bf16_rne(float f) {
    unsigned u = __float_as_uint(f);
    u += 0x7fffu + ((u >> 16) & 1u);
    return (unsigned short)(u >> 16);
}

// -------------------------- conversion kernels -----------------------------

__global__ void k_x_to_bf16(const float* __restrict__ in,
                            unsigned short* __restrict__ out, int n) {
    int i = blockIdx.x * blockDim.x + threadIdx.x;
    if (i < n) out[i] = f32_to_bf16_rne(in[i]);
}

// W [Fin, Fout] row-major f32 -> Wt [Fout, Fin] row-major bf16
__global__ void k_w_to_bf16_t(const float* __restrict__ W,
                              unsigned short* __restrict__ Wt,
                              int Fin, int Fout) {
    int i = blockIdx.x * blockDim.x + threadIdx.x;
    if (i >= Fin * Fout) return;
    int k = i / Fout, n = i % Fout;
    Wt[n * Fin + k] = f32_to_bf16_rne(W[i]);
}

// ------------------------------ WMMA GEMM ----------------------------------
// out[N,Fout] = X[N,Fin](bf16) @ Wt[Fout,Fin](bf16)^T, f32 accumulate.
// One wave = 16x32 output tile (two 16x16 accumulators), K-loop step 32.
// Fragment layouts used here (A and B share contiguous k = 16*half + s):
//   A 16x32: lanes 0-15 row M=lane, K=0..15; lanes 16-31 row
//            M=lane-16, K=16..31  -> two 16B loads
//   B 32x16: lanes 0-15 col N=lane, K=0..15;  lanes 16-31 col, K=16..31
//            -> one contiguous 32B load
//   C/D 16x16: VGPR r holds M=r (lanes 0-15) / M=8+r (lanes 16-31), N=lane&15
__global__ __launch_bounds__(256)
void k_gemm_bf16_wmma(const unsigned short* __restrict__ Xu,
                      const unsigned short* __restrict__ Wu,
                      float* __restrict__ out, int N, int Fin, int Fout) {
    const int wid  = threadIdx.x >> 5;
    const int lane = threadIdx.x & 31;
    const int tiles_n = Fout >> 5;            // 32-col tiles
    const int tiles_m = N >> 4;               // 16-row tiles
    int w = blockIdx.x * 8 + wid;
    if (w >= tiles_m * tiles_n) return;       // wave-uniform: EXEC stays full
    const int tm = w / tiles_n;
    const int tn = w % tiles_n;

    const __bf16* X = (const __bf16*)Xu;
    const __bf16* Wt = (const __bf16*)Wu;

    const int row    = (tm << 4) + (lane & 15);
    const int khalfA = (lane & 16) ? 16 : 0;  // A: K sub-offset per lane half
    const int koffB  = (lane & 16) ? 16 : 0;  // B: K sub-offset per lane half
    const int col0   = (tn << 5) + (lane & 15);

    const __bf16* arow  = X  + (size_t)row * Fin;
    const __bf16* b0row = Wt + (size_t)col0 * Fin;
    const __bf16* b1row = Wt + (size_t)(col0 + 16) * Fin;

    v8f acc0 = {};
    v8f acc1 = {};
    for (int k = 0; k < Fin; k += 32) {
        __builtin_prefetch(arow + k + 128, 0, 1);   // global_prefetch_b8
        v16bf a, b0, b1;
        *(v8bf*)&a        = *(const v8bf*)(arow + k + khalfA);
        *((v8bf*)&a + 1)  = *(const v8bf*)(arow + k + khalfA + 8);
        b0 = *(const v16bf*)(b0row + k + koffB);
        b1 = *(const v16bf*)(b1row + k + koffB);
        acc0 = __builtin_amdgcn_wmma_f32_16x16x32_bf16(
                   false, a, false, b0, (short)0, acc0, false, false);
        acc1 = __builtin_amdgcn_wmma_f32_16x16x32_bf16(
                   false, a, false, b1, (short)0, acc1, false, false);
    }

    const int rbase = (tm << 4) + ((lane & 16) ? 8 : 0);
    const int c0    = (tn << 5) + (lane & 15);
#pragma unroll
    for (int r = 0; r < 8; ++r) {
        out[(size_t)(rbase + r) * Fout + c0]      = acc0[r];
        out[(size_t)(rbase + r) * Fout + c0 + 16] = acc1[r];
    }
}

// -------------------------- attention kernels ------------------------------

// el/er = sum_d feat[n,h,d]*a{l,r}[h,d]; also init max-enc and denom.
__global__ void k_attn_coef(const float* __restrict__ feat,
                            const float* __restrict__ al,
                            const float* __restrict__ ar,
                            float* __restrict__ el, float* __restrict__ er,
                            unsigned* __restrict__ mEnc,
                            float* __restrict__ den,
                            int N, int H, int D) {
    int i = blockIdx.x * blockDim.x + threadIdx.x;   // over N*H
    if (i >= N * H) return;
    int h = i % H;
    const float* f = feat + (size_t)i * D;           // (n*H+h)*D
    float sl = 0.f, sr = 0.f;
    for (int d = 0; d < D; ++d) {
        float v = f[d];
        sl += v * al[h * D + d];
        sr += v * ar[h * D + d];
    }
    el[i] = sl;
    er[i] = sr;
    mEnc[i] = 0u;     // encodes "below every real value"
    den[i]  = 0.f;
}

__global__ void k_edge_max(const int* __restrict__ src, const int* __restrict__ dst,
                           const float* __restrict__ el, const float* __restrict__ er,
                           unsigned* __restrict__ mEnc, int E, int H) {
    int i = blockIdx.x * blockDim.x + threadIdx.x;   // over E*H
    if (i >= E * H) return;
    int e = i / H, h = i - e * H;
    int s = src[e], d = dst[e];
    float v = el[s * H + h] + er[d * H + h];
    v = v > 0.f ? v : 0.2f * v;                      // leaky_relu(0.2)
    atomicMax(&mEnc[d * H + h], enc_f32(v));
}

__global__ void k_edge_exp(const int* __restrict__ src, const int* __restrict__ dst,
                           const float* __restrict__ el, const float* __restrict__ er,
                           const unsigned* __restrict__ mEnc,
                           float* __restrict__ ee, float* __restrict__ den,
                           int E, int H) {
    int i = blockIdx.x * blockDim.x + threadIdx.x;   // over E*H
    if (i >= E * H) return;
    int e = i / H, h = i - e * H;
    int s = src[e], d = dst[e];
    float v = el[s * H + h] + er[d * H + h];
    v = v > 0.f ? v : 0.2f * v;
    float m = dec_f32(mEnc[d * H + h]);
    float x = expf(v - m);
    ee[i] = x;
    atomicAdd(&den[d * H + h], x);
}

__global__ void k_zero_f32(float* __restrict__ p, int n) {
    int i = blockIdx.x * blockDim.x + threadIdx.x;
    if (i < n) p[i] = 0.f;
}

// out[dst] += alpha * feat[src]; one thread = 4 floats of one edge message.
__global__ void k_aggregate(const int* __restrict__ src, const int* __restrict__ dst,
                            const float* __restrict__ feat,
                            const float* __restrict__ ee,
                            const float* __restrict__ den,
                            float* __restrict__ out,
                            int E, int H, int D, int F) {
    long long gid = (long long)blockIdx.x * blockDim.x + threadIdx.x;
    const int per = F >> 2;
    if (gid >= (long long)E * per) return;
    int e = (int)(gid / per);
    int j = (int)(gid - (long long)e * per) << 2;
    int h = j / D;
    int s = src[e], d = dst[e];
    float dn = den[d * H + h];
    dn = dn > 1e-9f ? dn : 1e-9f;
    float alpha = ee[(size_t)e * H + h] / dn;
    const float4 f = *(const float4*)(feat + (size_t)s * F + j);
    float* o = out + (size_t)d * F + j;
    atomicAdd(o + 0, alpha * f.x);
    atomicAdd(o + 1, alpha * f.y);
    atomicAdd(o + 2, alpha * f.z);
    atomicAdd(o + 3, alpha * f.w);
}

// bias + ELU, emit bf16 directly as next layer's GEMM input.
__global__ void k_bias_elu_bf16(const float* __restrict__ acc,
                                const float* __restrict__ b,
                                unsigned short* __restrict__ x16,
                                int N, int F) {
    int i = blockIdx.x * blockDim.x + threadIdx.x;   // over N*F
    if (i >= N * F) return;
    int j = i % F;
    float v = acc[i] + b[j];
    v = v > 0.f ? v : expm1f(v);                     // elu
    x16[i] = f32_to_bf16_rne(v);
}

// bias + mean over heads -> logits [N, C]
__global__ void k_final(const float* __restrict__ acc,
                        const float* __restrict__ b2,
                        float* __restrict__ out, int N, int H, int C) {
    int i = blockIdx.x * blockDim.x + threadIdx.x;   // over N*C
    if (i >= N * C) return;
    int n = i / C, c = i - n * C;
    float s = 0.f;
    for (int h = 0; h < H; ++h)
        s += acc[(size_t)n * H * C + h * C + c] + b2[h * C + c];
    out[i] = s * (1.0f / H);
}

// ------------------------------ host driver --------------------------------

static void gat_layer(const unsigned short* x16, const float* Wf,
                      const float* al, const float* ar,
                      unsigned short* wt16, float* feat, float* aggr,
                      float* el, float* er, unsigned* mEnc, float* den, float* ee,
                      const int* src, const int* dst,
                      int N, int E, int H, int D, int Fin, hipStream_t s) {
    const int Fout = H * D;
    const int nW = Fin * Fout;
    k_w_to_bf16_t<<<(nW + 255) / 256, 256, 0, s>>>(Wf, wt16, Fin, Fout);

    const int tiles = (N / 16) * (Fout / 32);
    k_gemm_bf16_wmma<<<(tiles + 7) / 8, 256, 0, s>>>(x16, wt16, feat, N, Fin, Fout);

    const int nNH = N * H;
    k_attn_coef<<<(nNH + 255) / 256, 256, 0, s>>>(feat, al, ar, el, er, mEnc, den, N, H, D);

    const int nEH = E * H;
    k_edge_max<<<(nEH + 255) / 256, 256, 0, s>>>(src, dst, el, er, mEnc, E, H);
    k_edge_exp<<<(nEH + 255) / 256, 256, 0, s>>>(src, dst, el, er, mEnc, ee, den, E, H);

    const int nNF = N * Fout;
    k_zero_f32<<<(nNF + 255) / 256, 256, 0, s>>>(aggr, nNF);

    const long long nAgg = (long long)E * (Fout / 4);
    k_aggregate<<<(unsigned)((nAgg + 255) / 256), 256, 0, s>>>(
        src, dst, feat, ee, den, aggr, E, H, D, Fout);
}

extern "C" void kernel_launch(void* const* d_in, const int* in_sizes, int n_in,
                              void* d_out, int out_size, void* d_ws, size_t ws_size,
                              hipStream_t stream) {
    const float* x   = (const float*)d_in[0];
    const int*   ei  = (const int*)d_in[1];
    const float* W0  = (const float*)d_in[2];
    const float* al0 = (const float*)d_in[3];
    const float* ar0 = (const float*)d_in[4];
    const float* b0  = (const float*)d_in[5];
    const float* W1  = (const float*)d_in[6];
    const float* al1 = (const float*)d_in[7];
    const float* ar1 = (const float*)d_in[8];
    const float* b1  = (const float*)d_in[9];
    const float* W2  = (const float*)d_in[10];
    const float* al2 = (const float*)d_in[11];
    const float* ar2 = (const float*)d_in[12];
    const float* b2  = (const float*)d_in[13];

    const int F_IN = 256, H = 4;
    const int N = in_sizes[0] / F_IN;        // 50000
    const int E = in_sizes[1] / 2;           // 800000
    const int D = in_sizes[3] / H;           // 64
    const int C = in_sizes[13] / H;          // 40
    const int F = H * D;                     // 256

    const int* src = ei;
    const int* dst = ei + E;

    char* ws = (char*)d_ws;
    size_t off = 0;
    auto take = [&](size_t bytes) -> void* {
        void* p = ws + off;
        off = (off + bytes + 255) & ~(size_t)255;
        return p;
    };
    float*          feat = (float*)take((size_t)N * F * 4);
    float*          aggr = (float*)take((size_t)N * F * 4);
    unsigned short* x16  = (unsigned short*)take((size_t)N * F * 2);
    unsigned short* wt16 = (unsigned short*)take((size_t)F_IN * F * 2);
    float*          el   = (float*)take((size_t)N * H * 4);
    float*          er   = (float*)take((size_t)N * H * 4);
    unsigned*       mEnc = (unsigned*)take((size_t)N * H * 4);
    float*          den  = (float*)take((size_t)N * H * 4);
    float*          ee   = (float*)take((size_t)E * H * 4);
    (void)ws_size; (void)n_in; (void)out_size;

    // Layer 0: input from d_in (f32 -> bf16), ELU.
    const int nX = N * F_IN;
    k_x_to_bf16<<<(nX + 255) / 256, 256, 0, stream>>>(x, x16, nX);
    gat_layer(x16, W0, al0, ar0, wt16, feat, aggr, el, er, mEnc, den, ee,
              src, dst, N, E, H, D, F_IN, stream);
    k_bias_elu_bf16<<<(N * F + 255) / 256, 256, 0, stream>>>(aggr, b0, x16, N, F);

    // Layer 1: ELU.
    gat_layer(x16, W1, al1, ar1, wt16, feat, aggr, el, er, mEnc, den, ee,
              src, dst, N, E, H, D, F, stream);
    k_bias_elu_bf16<<<(N * F + 255) / 256, 256, 0, stream>>>(aggr, b1, x16, N, F);

    // Layer 2: no activation; bias + head-mean into logits.
    gat_layer(x16, W2, al2, ar2, wt16, feat, aggr, el, er, mEnc, den, ee,
              src, dst, N, E, H, C, F, stream);
    k_final<<<(N * C + 255) / 256, 256, 0, stream>>>(aggr, b2, (float*)d_out, N, H, C);
}
